// TokenPooling_67379446940455
// MI455X (gfx1250) — compile-verified, matrix operand order checked by reading, not run on hardware
//
#include <hip/hip_runtime.h>
#include <math.h>

typedef __attribute__((ext_vector_type(16))) __bf16 v16bf;
typedef __attribute__((ext_vector_type(8)))  float  v8f;
typedef __attribute__((ext_vector_type(4)))  float  v4f;

// Problem constants (fixed by the reference)
constexpr int B_    = 16;
constexpr int LQ    = 64;
constexpr int LD    = 2048;
constexpr int H     = 1024;
constexpr int DIM   = 128;
constexpr int PF    = 4;
constexpr int MAXK  = LD / PF + 1;        // 513
constexpr int NT    = (MAXK + 15) / 16;   // 33 cluster tiles (padded to 528)
constexpr int ROWS_Q = B_ * LQ;           // 1024
constexpr int ROWS_D = B_ * LD;           // 32768
constexpr int ROWS   = ROWS_Q + ROWS_D;   // 33792
constexpr int KSTEPS = H / 32;            // 32 K-steps of 32 for the encode GEMM
constexpr int CTILES = DIM / 16;          // 8 column tiles of the projection

// Load 16 contiguous-ish fp32 (two float4 pairs) and convert to a bf16 A/B fragment slice.
__device__ inline v16bf load_cvt16(const float* __restrict__ p0,
                                   const float* __restrict__ p1) {
  v4f f0 = *(const v4f*)p0;
  v4f f1 = *(const v4f*)(p0 + 4);
  v4f f2 = *(const v4f*)p1;
  v4f f3 = *(const v4f*)(p1 + 4);
  v16bf a;
  a[0]=(__bf16)f0[0];  a[1]=(__bf16)f0[1];  a[2]=(__bf16)f0[2];  a[3]=(__bf16)f0[3];
  a[4]=(__bf16)f1[0];  a[5]=(__bf16)f1[1];  a[6]=(__bf16)f1[2];  a[7]=(__bf16)f1[3];
  a[8]=(__bf16)f2[0];  a[9]=(__bf16)f2[1];  a[10]=(__bf16)f2[2]; a[11]=(__bf16)f2[3];
  a[12]=(__bf16)f3[0]; a[13]=(__bf16)f3[1]; a[14]=(__bf16)f3[2]; a[15]=(__bf16)f3[3];
  return a;
}

// K-local element index -> (half, e) inverse of the 16-bit A/B fragment mapping:
//   e<8  : Klocal = half*8 + e          (K 0..15)
//   e>=8 : Klocal = 16 + half*8 + (e-8) (K 16..31)

// ---------------------------------------------------------------------------
// 1) Pack W [H,DIM] fp32 -> bf16 in B-fragment order: [s][c][lane][e]
// ---------------------------------------------------------------------------
__global__ void pack_w_kernel(const float* __restrict__ W,
                              unsigned short* __restrict__ Wp) {
  int idx = blockIdx.x * blockDim.x + threadIdx.x;
  if (idx >= KSTEPS * CTILES * 32 * 16) return;
  int e    = idx & 15;
  int lane = (idx >> 4) & 31;
  int c    = (idx >> 9) & 7;
  int s    = idx >> 12;
  int half = lane >> 4;
  int n    = c * 16 + (lane & 15);
  int kl   = (e < 8) ? (half * 8 + e) : (16 + half * 8 + (e - 8));
  int K    = s * 32 + kl;
  ((__bf16*)Wp)[idx] = (__bf16)W[(size_t)K * DIM + n];
}

// ---------------------------------------------------------------------------
// 2) Fused encode: r = hidden @ W + b ; L2-normalize rows ; * mask
//    One wave per 16-row tile; 8 f32 accumulators cover all DIM=128 columns.
// ---------------------------------------------------------------------------
__global__ void __launch_bounds__(128) encode_kernel(
    const float* __restrict__ qh, const float* __restrict__ dh,
    const unsigned short* __restrict__ Wp_u, const float* __restrict__ bias,
    const int* __restrict__ qmask, const int* __restrict__ dmask,
    float* __restrict__ qenc, float* __restrict__ denc) {
  const __bf16* __restrict__ Wp = (const __bf16*)Wp_u;
  int tile = blockIdx.x * 4 + (threadIdx.x >> 5);
  int lane = threadIdx.x & 31;
  int half = lane >> 4;
  int nl   = lane & 15;
  int row0 = tile * 16;
  bool isQ = row0 < ROWS_Q;
  int arow = row0 + nl;  // row this lane feeds into the A fragment
  const float* src = isQ ? (qh + (size_t)arow * H)
                         : (dh + (size_t)(arow - ROWS_Q) * H);

  v8f vzero = {0.f,0.f,0.f,0.f,0.f,0.f,0.f,0.f};
  v8f acc[8];
  #pragma unroll
  for (int c = 0; c < 8; ++c) acc[c] = vzero;

  for (int s = 0; s < KSTEPS; ++s) {
    const float* p = src + s * 32 + half * 8;
    v16bf a = load_cvt16(p, p + 16);
    const __bf16* wp = Wp + ((size_t)s * CTILES * 32 + lane) * 16;
    #pragma unroll
    for (int c = 0; c < 8; ++c) {
      v16bf bf = *(const v16bf*)(wp + (size_t)c * 32 * 16);
      acc[c] = __builtin_amdgcn_wmma_f32_16x16x32_bf16(
          false, a, false, bf, (short)0, acc[c], false, false);
    }
  }

  // Epilogue: bias, row sum-of-squares (reduce over n lanes), normalize, mask.
  float ss[8];
  #pragma unroll
  for (int r = 0; r < 8; ++r) ss[r] = 0.f;
  #pragma unroll
  for (int c = 0; c < 8; ++c) {
    float bn = bias[c * 16 + nl];
    #pragma unroll
    for (int r = 0; r < 8; ++r) {
      float v = acc[c][r] + bn;
      acc[c][r] = v;
      ss[r] += v * v;
    }
  }
  #pragma unroll
  for (int r = 0; r < 8; ++r) {
    #pragma unroll
    for (int w = 1; w < 16; w <<= 1)
      ss[r] += __shfl_xor(ss[r], w, 16);  // reduce across the 16 n-lanes of this half
  }
  #pragma unroll
  for (int r = 0; r < 8; ++r) {
    int m    = half * 8 + r;             // C/D layout row
    int grow = row0 + m;
    float mk = isQ ? (float)qmask[grow] : (float)dmask[grow - ROWS_Q];
    float nrm = fmaxf(sqrtf(ss[r]), 1e-12f);
    float sc  = mk / nrm;
    float* out = isQ ? (qenc + (size_t)grow * DIM)
                     : (denc + (size_t)(grow - ROWS_Q) * DIM);
    #pragma unroll
    for (int c = 0; c < 8; ++c) out[c * 16 + nl] = acc[c][r] * sc;
  }
}

// ---------------------------------------------------------------------------
// 3) Per-batch cluster count: k = max(sum(d_mask),2)/PF + 1
// ---------------------------------------------------------------------------
__global__ void kcount_kernel(const int* __restrict__ dmask, int* __restrict__ kArr) {
  __shared__ int sdata[256];
  int b = blockIdx.x, t = threadIdx.x;
  int sum = 0;
  for (int i = t; i < LD; i += 256) sum += dmask[b * LD + i];
  sdata[t] = sum; __syncthreads();
  for (int w = 128; w > 0; w >>= 1) {
    if (t < w) sdata[t] += sdata[t + w];
    __syncthreads();
  }
  if (t == 0) {
    int v = sdata[0];
    if (v < 2) v = 2;
    kArr[b] = v / PF + 1;
  }
}

// ---------------------------------------------------------------------------
// 4) Segment-sum pooling via global fp32 atomics (one block per token)
// ---------------------------------------------------------------------------
__global__ void pool_kernel(const float* __restrict__ denc,
                            const int* __restrict__ dmask,
                            const int* __restrict__ labels,
                            const int* __restrict__ kArr,
                            float* __restrict__ sums,
                            float* __restrict__ counts) {
  int tok = blockIdx.x;           // b*LD + t
  if (dmask[tok] == 0) return;
  int b   = tok / LD;
  int lab = labels[tok] % kArr[b];
  const float* src = denc + (size_t)tok * DIM;
  float* dst = sums + ((size_t)b * MAXK + lab) * DIM;
  atomicAdd(dst + threadIdx.x, src[threadIdx.x]);
  if (threadIdx.x == 0) atomicAdd(counts + (size_t)b * MAXK + lab, 1.0f);
}

// ---------------------------------------------------------------------------
// 5) Finalize pooled clusters: mean, normalize, pack to bf16 B-fragment order
//    Pp layout: [b][s(4)][ct(NT)][lane(32)][e(16)]
// ---------------------------------------------------------------------------
__global__ void finalize_kernel(const float* __restrict__ sums,
                                const float* __restrict__ counts,
                                unsigned short* __restrict__ Pp_u) {
  __shared__ float sh[128];
  int bc = blockIdx.x;            // b*MAXK + cl
  int b  = bc / MAXK, cl = bc % MAXK;
  int d  = threadIdx.x;           // 0..127
  float cnt  = fmaxf(counts[bc], 1.0f);
  float mean = sums[(size_t)bc * DIM + d] / cnt;
  sh[d] = mean * mean; __syncthreads();
  for (int w = 64; w > 0; w >>= 1) {
    if (d < w) sh[d] += sh[d + w];
    __syncthreads();
  }
  float val = mean / fmaxf(sqrtf(sh[0]), 1e-12f);
  int s  = d >> 5;
  int kl = d & 31;
  int half, e;
  if (kl < 8)       { half = 0; e = kl;      }
  else if (kl < 16) { half = 1; e = kl - 8;  }
  else if (kl < 24) { half = 0; e = kl - 8;  }
  else              { half = 1; e = kl - 16; }
  int lane = half * 16 + (cl & 15);
  int ct   = cl >> 4;
  size_t idx = ((((size_t)b * 4 + s) * NT + ct) * 32 + lane) * 16 + e;
  ((__bf16*)Pp_u)[idx] = (__bf16)val;
}

// ---------------------------------------------------------------------------
// 6) sim = q @ pooled^T via WMMA; masked row-max over clusters; q-masked sum.
//    One wave per (batch, 16-query tile); atomicAdd into out[b].
// ---------------------------------------------------------------------------
__global__ void __launch_bounds__(32) sim_kernel(
    const float* __restrict__ qenc, const unsigned short* __restrict__ Pp_u,
    const int* __restrict__ kArr, const int* __restrict__ qmask,
    float* __restrict__ out) {
  const __bf16* __restrict__ Pp = (const __bf16*)Pp_u;
  int b    = blockIdx.x >> 2;
  int qt   = blockIdx.x & 3;
  int lane = threadIdx.x & 31;
  int half = lane >> 4;
  int nl   = lane & 15;

  const float* src = qenc + (size_t)(b * LQ + qt * 16 + nl) * DIM;
  v16bf af[4];
  #pragma unroll
  for (int s = 0; s < 4; ++s) {
    const float* p = src + s * 32 + half * 8;
    af[s] = load_cvt16(p, p + 16);
  }
  int kb = kArr[b];
  v8f vzero = {0.f,0.f,0.f,0.f,0.f,0.f,0.f,0.f};
  float rmax[8];
  #pragma unroll
  for (int r = 0; r < 8; ++r) rmax[r] = -3.0e38f;

  for (int ct = 0; ct < NT; ++ct) {
    v8f acc = vzero;
    #pragma unroll
    for (int s = 0; s < 4; ++s) {
      const __bf16* bp = Pp + ((((size_t)b * 4 + s) * NT + ct) * 32 + lane) * 16;
      v16bf bf = *(const v16bf*)bp;
      acc = __builtin_amdgcn_wmma_f32_16x16x32_bf16(
          false, af[s], false, bf, (short)0, acc, false, false);
    }
    bool valid = (ct * 16 + nl) < kb;   // pad_mask: cluster index < k
    #pragma unroll
    for (int r = 0; r < 8; ++r) {
      float cur = valid ? acc[r] : -1e4f;
      rmax[r] = fmaxf(rmax[r], cur);
    }
  }
  #pragma unroll
  for (int r = 0; r < 8; ++r) {
    #pragma unroll
    for (int w = 1; w < 16; w <<= 1)
      rmax[r] = fmaxf(rmax[r], __shfl_xor(rmax[r], w, 16));
  }
  float total = 0.f;
  #pragma unroll
  for (int r = 0; r < 8; ++r) {
    int grow = b * LQ + qt * 16 + half * 8 + r;
    total += rmax[r] * (float)qmask[grow];
  }
  if (nl == 0) atomicAdd(out + b, total);  // two contributing lanes per wave (halves)
}

// ---------------------------------------------------------------------------
extern "C" void kernel_launch(void* const* d_in, const int* in_sizes, int n_in,
                              void* d_out, int out_size, void* d_ws, size_t ws_size,
                              hipStream_t stream) {
  (void)in_sizes; (void)n_in; (void)out_size; (void)ws_size;
  const float* qh     = (const float*)d_in[0];
  const float* dh     = (const float*)d_in[1];
  const float* W      = (const float*)d_in[2];
  const float* bias   = (const float*)d_in[3];
  const int*   qmask  = (const int*)d_in[4];
  const int*   dmask  = (const int*)d_in[5];
  const int*   labels = (const int*)d_in[6];

  char* base = (char*)d_ws;
  size_t off = 0;
  auto alloc = [&](size_t bytes) {
    size_t o = off;
    off += (bytes + 255) & ~(size_t)255;
    return o;
  };
  unsigned short* Wp  = (unsigned short*)(base + alloc((size_t)KSTEPS * CTILES * 32 * 16 * 2));
  float* qenc   = (float*)(base + alloc((size_t)ROWS_Q * DIM * sizeof(float)));
  float* denc   = (float*)(base + alloc((size_t)ROWS_D * DIM * sizeof(float)));
  float* sums   = (float*)(base + alloc((size_t)B_ * MAXK * DIM * sizeof(float)));
  float* counts = (float*)(base + alloc((size_t)B_ * MAXK * sizeof(float)));
  int*   kArr   = (int*)(base + alloc((size_t)B_ * sizeof(int)));
  unsigned short* Pp = (unsigned short*)(base + alloc((size_t)B_ * 4 * NT * 32 * 16 * 2));
  float* out = (float*)d_out;

  hipMemsetAsync(sums,   0, (size_t)B_ * MAXK * DIM * sizeof(float), stream);
  hipMemsetAsync(counts, 0, (size_t)B_ * MAXK * sizeof(float), stream);
  hipMemsetAsync(Pp,     0, (size_t)B_ * 4 * NT * 32 * 16 * 2, stream);
  hipMemsetAsync(out,    0, (size_t)B_ * sizeof(float), stream);

  pack_w_kernel<<<(KSTEPS * CTILES * 32 * 16 + 255) / 256, 256, 0, stream>>>(W, Wp);
  encode_kernel<<<(ROWS / 16) / 4, 128, 0, stream>>>(qh, dh, Wp, bias, qmask, dmask,
                                                     qenc, denc);
  kcount_kernel<<<B_, 256, 0, stream>>>(dmask, kArr);
  pool_kernel<<<B_ * LD, DIM, 0, stream>>>(denc, dmask, labels, kArr, sums, counts);
  finalize_kernel<<<B_ * MAXK, DIM, 0, stream>>>(sums, counts, Pp);
  sim_kernel<<<B_ * 4, 32, 0, stream>>>(qenc, Pp, kArr, qmask, out);
}
